// Structure_aware_THTN2_16552803959367
// MI455X (gfx1250) — matrix operand order repaired and checked by reading, not compiled, not run on hardware
//
#include <hip/hip_runtime.h>
#include <hip/hip_bf16.h>
#include <math.h>

#define NN 100000   // nodes
#define NE 10000    // hyperedges
#define NM 1000000  // messages
#define DD 128
#define NCLS 40
#define SCALE 0.08838834764831845f  // 1/sqrt(128)

typedef __attribute__((ext_vector_type(16))) _Float16 v16h;
typedef __attribute__((ext_vector_type(8)))  float    v8f;

// ---------------------------------------------------------------------------
// Weight prep: convert f32 W[K=128, Do=128] -> f16 Wt[Do=128, K=128] (transposed)
// so a WMMA B-fragment column is contiguous.
// ---------------------------------------------------------------------------
__global__ __launch_bounds__(256) void convtrans128(const float* __restrict__ W,
                                                    _Float16* __restrict__ Wt) {
  int i = blockIdx.x * 256 + threadIdx.x;          // 0..16383
  int k = i >> 7, n = i & 127;
  Wt[n * 128 + k] = (_Float16)W[k * 128 + n];
}

__global__ __launch_bounds__(256) void fill_f32(float* __restrict__ p, float v, int n) {
  int i = blockIdx.x * 256 + threadIdx.x;
  if (i < n) p[i] = v;
}

// ---------------------------------------------------------------------------
// Y[R,128] = act( X[R,128] @ W[128,128] + bias (+ addres) ), WMMA f16->f32.
// Block = 256 threads = 8 waves; each wave owns one 16x16 output tile,
// block covers 16 rows x 128 cols. R must be a multiple of 16 (N, E are).
// ---------------------------------------------------------------------------
__global__ __launch_bounds__(256)
void gemm128_wmma(const float* __restrict__ X, const _Float16* __restrict__ Wt,
                  const float* __restrict__ bias, const float* __restrict__ addres,
                  float* __restrict__ Y, int R, int doRelu) {
  const int lane = threadIdx.x & 31;
  const int wid  = threadIdx.x >> 5;      // 0..7 -> column tile
  const int rowBase = blockIdx.x * 16;
  const int n0 = wid * 16;

  const int mrow = rowBase + (lane & 15);
  const int kb0  = (lane & 16) ? 16 : 0;  // A-fragment K sub-offset (contiguous k = 16h + s, matches B)
  const int nB   = n0 + (lane & 15);
  const int kbB  = (lane & 16) ? 16 : 0;  // B-fragment K sub-offset (contiguous k = 16h + s, matches A)

  const float*    xrow = X  + (size_t)mrow * 128;
  const _Float16* wcol = Wt + (size_t)nB   * 128;

  v8f acc = {};
#pragma unroll
  for (int kc = 0; kc < 128; kc += 32) {
    float4 x0 = *(const float4*)(xrow + kc + kb0);
    float4 x1 = *(const float4*)(xrow + kc + kb0 + 4);
    float4 x2 = *(const float4*)(xrow + kc + kb0 + 8);
    float4 x3 = *(const float4*)(xrow + kc + kb0 + 12);
    v16h a;
    a[0]=(_Float16)x0.x; a[1]=(_Float16)x0.y; a[2] =(_Float16)x0.z; a[3] =(_Float16)x0.w;
    a[4]=(_Float16)x1.x; a[5]=(_Float16)x1.y; a[6] =(_Float16)x1.z; a[7] =(_Float16)x1.w;
    a[8]=(_Float16)x2.x; a[9]=(_Float16)x2.y; a[10]=(_Float16)x2.z; a[11]=(_Float16)x2.w;
    a[12]=(_Float16)x3.x; a[13]=(_Float16)x3.y; a[14]=(_Float16)x3.z; a[15]=(_Float16)x3.w;
    v16h b = *(const v16h*)(wcol + kc + kbB);   // 32B contiguous column slice
    acc = __builtin_amdgcn_wmma_f32_16x16x32_f16(false, a, false, b, (short)0, acc,
                                                 false, false);
  }

  const int mOff = (lane & 16) ? 8 : 0;
  const int n = n0 + (lane & 15);
#pragma unroll
  for (int r = 0; r < 8; ++r) {
    int m = rowBase + r + mOff;
    float v = acc[r] + bias[n];
    if (addres) v += addres[(size_t)m * 128 + n];
    if (doRelu) v = v > 0.f ? v : 0.f;
    Y[(size_t)m * 128 + n] = v;
  }
}

// fv[row,:] += cs_emb[ci[row],:] + un_emb[ui[row],:]   (wave per row)
__global__ __launch_bounds__(256)
void add_emb(float* __restrict__ fv, const float* __restrict__ cs,
             const float* __restrict__ un, const int* __restrict__ ci,
             const int* __restrict__ ui, int R) {
  int row = blockIdx.x * 8 + (threadIdx.x >> 5);
  if (row >= R) return;
  int lane = threadIdx.x & 31;
  float4* f = (float4*)(fv + (size_t)row * 128) + lane;
  float4 a = *((const float4*)(cs + (size_t)ci[row] * 128) + lane);
  float4 b = *((const float4*)(un + (size_t)ui[row] * 128) + lane);
  float4 x = *f;
  x.x += a.x + b.x; x.y += a.y + b.y; x.z += a.z + b.z; x.w += a.w + b.w;
  *f = x;
}

__device__ inline float wave_sum32(float v) {
#pragma unroll
  for (int m = 16; m >= 1; m >>= 1) v += __shfl_xor(v, m, 32);
  return v;
}

// Y = LN(Xin (+ resid)) * g + b   (wave per row of 128)
__global__ __launch_bounds__(256)
void ln128(const float* __restrict__ Xin, const float* __restrict__ resid,
           const float* __restrict__ g, const float* __restrict__ b,
           float* __restrict__ Y, int R) {
  int row = blockIdx.x * 8 + (threadIdx.x >> 5);
  if (row >= R) return;
  int lane = threadIdx.x & 31;
  float4 x = *((const float4*)(Xin + (size_t)row * 128) + lane);
  if (resid) {
    float4 r0 = *((const float4*)(resid + (size_t)row * 128) + lane);
    x.x += r0.x; x.y += r0.y; x.z += r0.z; x.w += r0.w;
  }
  float mu = wave_sum32(x.x + x.y + x.z + x.w) * (1.f / 128.f);
  float d0 = x.x - mu, d1 = x.y - mu, d2 = x.z - mu, d3 = x.w - mu;
  float var = wave_sum32(d0 * d0 + d1 * d1 + d2 * d2 + d3 * d3) * (1.f / 128.f);
  float rstd = rsqrtf(var + 1e-5f);
  float4 gg = *((const float4*)g + lane);
  float4 bb = *((const float4*)b + lane);
  float4 o;
  o.x = d0 * rstd * gg.x + bb.x; o.y = d1 * rstd * gg.y + bb.y;
  o.z = d2 * rstd * gg.z + bb.z; o.w = d3 * rstd * gg.w + bb.w;
  *((float4*)(Y + (size_t)row * 128) + lane) = o;
}

// s[m] = leaky(dot(K[idxK[m]], Q[idxSeg[m]])) * SCALE + bias[m]; segmax[idxSeg[m]] = max
__global__ __launch_bounds__(256)
void score_pass(const float* __restrict__ K, const int* __restrict__ idxK,
                const float* __restrict__ Q, const int* __restrict__ idxSeg,
                const float* __restrict__ bias, float* __restrict__ S,
                float* __restrict__ segmax, int M) {
  int m = blockIdx.x * 256 + threadIdx.x;
  if (m >= M) return;
  const float4* kr = (const float4*)(K + (size_t)idxK[m] * 128);
  const float4* qr = (const float4*)(Q + (size_t)idxSeg[m] * 128);
  float acc = 0.f;
#pragma unroll
  for (int i = 0; i < 32; ++i) {
    float4 a = kr[i], b = qr[i];
    acc += a.x * b.x + a.y * b.y + a.z * b.z + a.w * b.w;
  }
  float s = (acc > 0.f ? acc : 0.01f * acc) * SCALE + bias[m];
  S[m] = s;
  atomicMax(segmax + idxSeg[m], s);    // f32 max atomic -> global_atomic_max_num_f32
}

// S[m] = exp(S[m] - segmax[seg]); den[seg] += S[m]
__global__ __launch_bounds__(256)
void exp_pass(float* __restrict__ S, const float* __restrict__ segmax,
              const int* __restrict__ idxSeg, float* __restrict__ den, int M) {
  int m = blockIdx.x * 256 + threadIdx.x;
  if (m >= M) return;
  int sg = idxSeg[m];
  float e = __expf(S[m] - segmax[sg]);
  S[m] = e;
  atomicAdd(den + sg, e);
}

// H[seg,:] += (S[m]/den[seg]) * V[idxV[m],:]   (wave per message, 4 floats/lane)
__global__ __launch_bounds__(256)
void agg_pass(const float* __restrict__ V, const int* __restrict__ idxV,
              const int* __restrict__ idxSeg, const float* __restrict__ S,
              const float* __restrict__ den, float* __restrict__ H, int M) {
  int m = blockIdx.x * 8 + (threadIdx.x >> 5);
  if (m >= M) return;
  int lane = threadIdx.x & 31;
  int sg = idxSeg[m];
  float w = S[m] / den[sg];
  float4 v = *((const float4*)(V + (size_t)idxV[m] * 128) + lane);
  float* h = H + (size_t)sg * 128 + lane * 4;
  atomicAdd(h + 0, w * v.x); atomicAdd(h + 1, w * v.y);
  atomicAdd(h + 2, w * v.z); atomicAdd(h + 3, w * v.w);
}

// pred[row,c] = dot(X[row], W_cls[:,c]) + b_cls[c]   (Do=40, scalar path)
__global__ __launch_bounds__(256)
void cls_gemm(const float* __restrict__ X, const float* __restrict__ W,
              const float* __restrict__ b, float* __restrict__ Y, int R) {
  int i = blockIdx.x * 256 + threadIdx.x;
  if (i >= R * NCLS) return;
  int row = i / NCLS, c = i - row * NCLS;
  const float* x = X + (size_t)row * 128;
  float acc = b[c];
#pragma unroll 8
  for (int k = 0; k < 128; ++k) acc += x[k] * W[k * NCLS + c];
  Y[i] = acc;
}

// ---------------------------------------------------------------------------
static inline size_t al256(size_t x) { return (x + 255) & ~(size_t)255; }

extern "C" void kernel_launch(void* const* d_in, const int* in_sizes, int n_in,
                              void* d_out, int out_size, void* d_ws, size_t ws_size,
                              hipStream_t stream) {
  (void)in_sizes; (void)n_in; (void)out_size; (void)ws_size;
  const float* vfeat    = (const float*)d_in[0];
  const float* efeat    = (const float*)d_in[1];
  const float* bias_in  = (const float*)d_in[2];
  const float* bias_con = (const float*)d_in[3];
  const int*   cidx     = (const int*)d_in[4];
  const int*   uidx     = (const int*)d_in[5];
  const int*   node_idx = (const int*)d_in[6];
  const int*   edge_idx = (const int*)d_in[7];
  const float* W_vtx1 = (const float*)d_in[8];   const float* b_vtx1 = (const float*)d_in[9];
  const float* cs_emb = (const float*)d_in[10];  const float* un_emb = (const float*)d_in[11];
  const float* W_qe = (const float*)d_in[12];    const float* b_qe = (const float*)d_in[13];
  const float* W_kv = (const float*)d_in[14];    const float* b_kv = (const float*)d_in[15];
  const float* W_vv = (const float*)d_in[16];    const float* b_vv = (const float*)d_in[17];
  const float* W_qv = (const float*)d_in[18];    const float* b_qv = (const float*)d_in[19];
  const float* W_ke = (const float*)d_in[20];    const float* b_ke = (const float*)d_in[21];
  const float* W_ve = (const float*)d_in[22];    const float* b_ve = (const float*)d_in[23];
  const float* ln1g = (const float*)d_in[24];    const float* ln1b = (const float*)d_in[25];
  const float* ln2g = (const float*)d_in[26];    const float* ln2b = (const float*)d_in[27];
  const float* W_l1 = (const float*)d_in[28];    const float* b_l1 = (const float*)d_in[29];
  const float* W_l2 = (const float*)d_in[30];    const float* b_l2 = (const float*)d_in[31];
  const float* W_l3 = (const float*)d_in[32];    const float* b_l3 = (const float*)d_in[33];
  const float* W_l4 = (const float*)d_in[34];    const float* b_l4 = (const float*)d_in[35];
  const float* W_cls= (const float*)d_in[36];    const float* b_cls= (const float*)d_in[37];
  const float* W_mlp= (const float*)d_in[38];    const float* b_mlp= (const float*)d_in[39];

  // ---- workspace layout ----
  char* ws = (char*)d_ws; size_t off = 0;
  auto alloc = [&](size_t bytes) -> char* { char* p = ws + off; off = al256(off + bytes); return p; };
  _Float16* wt  = (_Float16*)alloc(12 * 16384 * sizeof(_Float16));  // 12 transposed f16 weights
  float* fv  = (float*)alloc((size_t)NN * 128 * 4);
  float* kv  = (float*)alloc((size_t)NN * 128 * 4);   // reused: hV, then pre-LN tmp
  float* vv  = (float*)alloc((size_t)NN * 128 * 4);   // reused: yv
  float* qv  = (float*)alloc((size_t)NN * 128 * 4);   // reused: ffn2 hidden
  float* qe  = (float*)alloc((size_t)NE * 128 * 4);
  float* hE  = (float*)alloc((size_t)NE * 128 * 4);   // reused: pre-LN tmp (edge)
  float* xe  = (float*)alloc((size_t)NE * 128 * 4);
  float* te  = (float*)alloc((size_t)NE * 128 * 4);
  float* fe2 = (float*)alloc((size_t)NE * 128 * 4);
  float* ke  = (float*)alloc((size_t)NE * 128 * 4);
  float* ve  = (float*)alloc((size_t)NE * 128 * 4);
  float* sbuf = (float*)alloc((size_t)NM * 4);
  float* maxE = (float*)alloc((size_t)NE * 4);
  float* denE = (float*)alloc((size_t)NE * 4);
  float* maxN = (float*)alloc((size_t)NN * 4);
  float* denN = (float*)alloc((size_t)NN * 4);

  float* pred  = (float*)d_out;
  float* fvout = (float*)d_out + (size_t)NN * NCLS;

  _Float16* WTvtx1 = wt + 0*16384;  _Float16* WTkv = wt + 1*16384;
  _Float16* WTvv   = wt + 2*16384;  _Float16* WTqv = wt + 3*16384;
  _Float16* WTqe   = wt + 4*16384;  _Float16* WTke = wt + 5*16384;
  _Float16* WTve   = wt + 6*16384;  _Float16* WTl1 = wt + 7*16384;
  _Float16* WTl2   = wt + 8*16384;  _Float16* WTl3 = wt + 9*16384;
  _Float16* WTl4   = wt + 10*16384; _Float16* WTmlp= wt + 11*16384;

  // ---- weight prep ----
  convtrans128<<<64, 256, 0, stream>>>(W_vtx1, WTvtx1);
  convtrans128<<<64, 256, 0, stream>>>(W_kv, WTkv);
  convtrans128<<<64, 256, 0, stream>>>(W_vv, WTvv);
  convtrans128<<<64, 256, 0, stream>>>(W_qv, WTqv);
  convtrans128<<<64, 256, 0, stream>>>(W_qe, WTqe);
  convtrans128<<<64, 256, 0, stream>>>(W_ke, WTke);
  convtrans128<<<64, 256, 0, stream>>>(W_ve, WTve);
  convtrans128<<<64, 256, 0, stream>>>(W_l1, WTl1);
  convtrans128<<<64, 256, 0, stream>>>(W_l2, WTl2);
  convtrans128<<<64, 256, 0, stream>>>(W_l3, WTl3);
  convtrans128<<<64, 256, 0, stream>>>(W_l4, WTl4);
  convtrans128<<<64, 256, 0, stream>>>(W_mlp, WTmlp);

  // ---- segment-stat init ----
  fill_f32<<<(NE + 255) / 256, 256, 0, stream>>>(maxE, -INFINITY, NE);
  fill_f32<<<(NE + 255) / 256, 256, 0, stream>>>(denE, 0.f, NE);
  fill_f32<<<((NE * 128) + 255) / 256, 256, 0, stream>>>(hE, 0.f, NE * 128);
  fill_f32<<<(NN + 255) / 256, 256, 0, stream>>>(maxN, -INFINITY, NN);
  fill_f32<<<(NN + 255) / 256, 256, 0, stream>>>(denN, 0.f, NN);

  // ---- feat_v = vfeat @ W_vtx1 + b + cs_emb[ci] + un_emb[ui] ----
  gemm128_wmma<<<NN / 16, 256, 0, stream>>>(vfeat, WTvtx1, b_vtx1, nullptr, fv, NN, 0);
  add_emb<<<(NN + 7) / 8, 256, 0, stream>>>(fv, cs_emb, un_emb, cidx, uidx, NN);

  // ---- projections ----
  gemm128_wmma<<<NN / 16, 256, 0, stream>>>(fv, WTkv, b_kv, nullptr, kv, NN, 0);
  gemm128_wmma<<<NN / 16, 256, 0, stream>>>(fv, WTvv, b_vv, nullptr, vv, NN, 0);
  gemm128_wmma<<<NN / 16, 256, 0, stream>>>(fv, WTqv, b_qv, nullptr, qv, NN, 0);
  gemm128_wmma<<<NE / 16, 256, 0, stream>>>(efeat, WTqe, b_qe, nullptr, qe, NE, 0);

  // ---- stage 1: node -> edge attention (segments = edge_idx) ----
  score_pass<<<(NM + 255) / 256, 256, 0, stream>>>(kv, node_idx, qe, edge_idx, bias_in, sbuf, maxE, NM);
  exp_pass  <<<(NM + 255) / 256, 256, 0, stream>>>(sbuf, maxE, edge_idx, denE, NM);
  agg_pass  <<<(NM + 7) / 8, 256, 0, stream>>>(vv, node_idx, edge_idx, sbuf, denE, hE, NM);

  // x = LN(h_e + feat_e); ffn; feat_e' = LN(ffn + x)
  ln128<<<(NE + 7) / 8, 256, 0, stream>>>(hE, efeat, ln1g, ln1b, xe, NE);
  gemm128_wmma<<<NE / 16, 256, 0, stream>>>(xe, WTl1, b_l1, nullptr, te, NE, 1);
  gemm128_wmma<<<NE / 16, 256, 0, stream>>>(te, WTl2, b_l2, xe, hE, NE, 0);
  ln128<<<(NE + 7) / 8, 256, 0, stream>>>(hE, nullptr, ln1g, ln1b, fe2, NE);

  // ---- stage 2: edge -> node attention (segments = node_idx) ----
  gemm128_wmma<<<NE / 16, 256, 0, stream>>>(fe2, WTke, b_ke, nullptr, ke, NE, 0);
  gemm128_wmma<<<NE / 16, 256, 0, stream>>>(fe2, WTve, b_ve, nullptr, ve, NE, 0);

  float* hV = kv;  // kv dead after stage-1 scores
  fill_f32<<<(((size_t)NN * 128) + 255) / 256, 256, 0, stream>>>(hV, 0.f, NN * 128);
  score_pass<<<(NM + 255) / 256, 256, 0, stream>>>(ke, edge_idx, qv, node_idx, bias_con, sbuf, maxN, NM);
  exp_pass  <<<(NM + 255) / 256, 256, 0, stream>>>(sbuf, maxN, node_idx, denN, NM);
  agg_pass  <<<(NM + 7) / 8, 256, 0, stream>>>(ve, edge_idx, node_idx, sbuf, denN, hV, NM);

  // y = LN(h_v + feat_v); ffn2; feat_v' = LN(ffn2 + y)
  float* yv = vv;                 // vv dead after stage-1 aggregation
  float* t1 = qv;                 // qv dead after stage-2 scores
  ln128<<<(NN + 7) / 8, 256, 0, stream>>>(hV, fv, ln2g, ln2b, yv, NN);
  float* tmpN = hV;               // hV dead after LN above
  gemm128_wmma<<<NN / 16, 256, 0, stream>>>(yv, WTl3, b_l3, nullptr, t1, NN, 1);
  gemm128_wmma<<<NN / 16, 256, 0, stream>>>(t1, WTl4, b_l4, yv, tmpN, NN, 0);
  float* fvf = fv;                // fv dead after residual into yv
  ln128<<<(NN + 7) / 8, 256, 0, stream>>>(tmpN, nullptr, ln2g, ln2b, fvf, NN);

  // ---- heads ----
  cls_gemm<<<((NN * NCLS) + 255) / 256, 256, 0, stream>>>(fvf, W_cls, b_cls, pred, NN);
  gemm128_wmma<<<NN / 16, 256, 0, stream>>>(fvf, WTmlp, b_mlp, nullptr, fvout, NN, 0);
}
